// MultiHeadAttention_37941741093164
// MI455X (gfx1250) — matrix operand order repaired and checked by reading, not compiled, not run on hardware
//
#include <hip/hip_runtime.h>
#include <hip/hip_bf16.h>
#include <stdint.h>

// ---------------------------------------------------------------------------
// MI455X / gfx1250 multi-head attention, bf16 WMMA path + TDM async tiles.
// B=2, S=2048, D=1024, H=16, DH=64.  Softmax is over the QUERY axis (axis=2)
// per the reference -> per-key-column normalization, two-pass scheme.
// Q pre-scaled by (1/sqrt(1024))*log2(e) => scores live in the exp2 domain.
// Mask bit-packed+transposed (1 u32 per lane per 16x16 tile).
// Pass 2 K/V tiles are DMA'd into double-buffered LDS by the Tensor Data
// Mover (tensor_load_to_lds + s_wait_tensorcnt), shared by all 4 waves.
// ---------------------------------------------------------------------------

#define Bb  2
#define Ss  2048
#define Dd  1024
#define Hh  16
#define DHh 64

typedef __attribute__((ext_vector_type(16))) __bf16    v16bf;
typedef __attribute__((ext_vector_type(8)))  float     v8f;
typedef __attribute__((ext_vector_type(4)))  uint32_t  u32x4;
typedef __attribute__((ext_vector_type(8)))  int       i32x8;
typedef __attribute__((ext_vector_type(4)))  int       i32x4;

union FragBF {             // one WMMA 16x32 bf16 operand: 16 bf16 per lane
  uint4 u4[2];
  v16bf v;
};

__device__ __forceinline__ uint16_t f2bf(float f) {   // fp32 -> bf16 RNE
  uint32_t u = __float_as_uint(f);
  u += 0x7FFFu + ((u >> 16) & 1u);
  return (uint16_t)(u >> 16);
}

__device__ __forceinline__ v8f wmma_bf16(const FragBF& a, const FragBF& b, v8f c) {
  return __builtin_amdgcn_wmma_f32_16x16x32_bf16(false, a.v, false, b.v,
                                                 (short)0, c, false, false);
}

// 1/sqrt(1024) * log2(e): folded into Qbf so WMMA scores are exp2-ready.
#define SCL_LOG2E 0.04508422f

// ---------------------------------------------------------------------------
// TDM: issue a 2-D bf16 tile load (tile_d1 rows x tile_d0 elems, row stride
// d0_stride elems) from global into LDS.  Descriptor per CDNA5 ISA 8.3/8.4.
// ---------------------------------------------------------------------------
__device__ __forceinline__ uint32_t lds_off(const void* p) {
  // generic pointer to LDS: addr[31:0] is the LDS byte offset (ISA 10.2)
  return (uint32_t)(uintptr_t)p;
}

__device__ __forceinline__ void tdm_load_2d(uint32_t ldsoff, const void* gptr,
                                            uint32_t tensor_d0, uint32_t tensor_d1,
                                            uint32_t tile_d0, uint32_t tile_d1,
                                            uint32_t d0_stride) {
  uint64_t ga = (uint64_t)(uintptr_t)gptr;
  u32x4 g0;
  g0[0] = 1u;                                            // count=1 (valid D#)
  g0[1] = ldsoff;                                        // lds_addr (bytes)
  g0[2] = (uint32_t)ga;                                  // global_addr[31:0]
  g0[3] = (uint32_t)((ga >> 32) & 0x1FFFFFFu) | (2u << 30);  // [56:32] | type=2
  i32x8 g1;
  g1[0] = (int)(1u << 16);                               // data_size=1 (2 bytes)
  g1[1] = (int)(tensor_d0 << 16);                        // tensor_dim0[15:0]
  g1[2] = (int)((tensor_d0 >> 16) | (tensor_d1 << 16));  // dim0[31:16]|dim1[15:0]
  g1[3] = (int)((tensor_d1 >> 16) | (tile_d0 << 16));    // dim1[31:16]|tile_dim0
  g1[4] = (int)tile_d1;                                  // tile_dim1 (tile_dim2=0)
  g1[5] = (int)d0_stride;                                // tensor_dim0_stride[31:0]
  g1[6] = 0;
  g1[7] = 0;
  i32x4 z4 = {0, 0, 0, 0};
#if defined(__clang_major__) && __clang_major__ >= 23
  i32x8 z8 = {0, 0, 0, 0, 0, 0, 0, 0};
  __builtin_amdgcn_tensor_load_to_lds(g0, g1, z4, z4, z8, 0);
#else
  __builtin_amdgcn_tensor_load_to_lds(g0, g1, z4, z4, 0);
#endif
}

// ---------------------------------------------------------------------------
// Bit-pack + transpose the bool mask: maskT[(b*64 + q/32)*2048 + k] bit(q&31)
// = mask[b][q][k].  1MB total -> L2 resident for both attention passes.
// ---------------------------------------------------------------------------
__global__ __launch_bounds__(256) void mask_pack(const uint8_t* __restrict__ mask,
                                                 uint32_t* __restrict__ maskT) {
  int tid = blockIdx.x * 256 + threadIdx.x;   // 0 .. 2*64*2048-1
  int k  = tid & (Ss - 1);
  int qw = (tid >> 11) & 63;
  int b  = tid >> 17;
  const uint8_t* mp = mask + ((size_t)b * Ss + qw * 32) * Ss + k;
  uint32_t w = 0;
#pragma unroll
  for (int j = 0; j < 32; ++j)
    w |= (uint32_t)(mp[(size_t)j * Ss] & 1) << j;
  maskT[tid] = w;
}

// ---------------------------------------------------------------------------
// Generic [4096 x 1024] x [1024 x 1024] GEMM, fp32 in, bf16 WMMA, f32 acc.
// 256 threads = 8 waves; block tile 128x128; wave tile 32x64 (2x4 WMMA tiles).
// A/B staged through LDS in exact WMMA fragment order (lane-contiguous 32B).
// ---------------------------------------------------------------------------
enum { MODE_Q = 0, MODE_K = 1, MODE_V = 2, MODE_O = 3 };

template <int MODE>
__global__ __launch_bounds__(256) void gemm_proj(
    const float* __restrict__ A, const float* __restrict__ W,
    float* __restrict__ outF, uint16_t* __restrict__ outB) {
  __shared__ uint16_t lA[8 * 512];  // 8 M-subtiles, fragment order, 8KB
  __shared__ uint16_t lB[8 * 512];  // 8 N-subtiles, fragment order, 8KB

  const int tid  = threadIdx.x;
  const int lane = tid & 31;
  const int wv   = tid >> 5;
  const int wm   = wv >> 1;          // 0..3  (wave row)
  const int wn   = wv & 1;           // 0..1  (wave col)
  const int hi   = lane >> 4;
  const int ln   = lane & 15;
  const int m0   = blockIdx.x * 128;
  const int n0   = blockIdx.y * 128;

  const v8f vzero = {0, 0, 0, 0, 0, 0, 0, 0};
  v8f acc[2][4];
#pragma unroll
  for (int i = 0; i < 2; ++i)
#pragma unroll
    for (int n = 0; n < 4; ++n) acc[i][n] = vzero;

  uint32_t* lA32 = (uint32_t*)lA;

  for (int k0 = 0; k0 < 1024; k0 += 32) {
    // ---- stage A tile [128 x 32] fp32 -> bf16 fragment order -------------
#pragma unroll
    for (int i = 0; i < 8; ++i) {
      int e = tid + i * 256;                  // 0..2047 (m-row, kk-pair)
      int kp = e & 15, row = (e >> 4) & 15, mt = e >> 8;
      int m = m0 + mt * 16 + row;
      int kk = kp * 2;
      float2 f = *(const float2*)(A + (size_t)m * 1024 + k0 + kk);
      int g = kk >> 4, h2 = (kk >> 3) & 1, jj = (kk & 7) >> 1;
      int l = row + h2 * 16;
      int j = g * 4 + jj;
      uint32_t packed = (uint32_t)f2bf(f.x) | ((uint32_t)f2bf(f.y) << 16);
      lA32[mt * 256 + l * 8 + j] = packed;
    }
    // ---- stage B tile [32 x 128] fp32 -> bf16 fragment order -------------
#pragma unroll
    for (int i = 0; i < 8; ++i) {
      int e = tid + i * 256;                  // 0..2047 (K-row, n-pair)
      int np = e & 63, K = e >> 6;            // K: 0..31
      int n = np * 2;
      float2 f = *(const float2*)(W + (size_t)(k0 + K) * 1024 + n0 + n);
      int h2 = (K >> 3) & 1, j = ((K >> 4) * 8 + (K & 7)) >> 1, p = K & 1;
      int nt = n >> 4;                         // n, n+1 share the subtile
      lB[nt * 512 + ((n & 15) + h2 * 16) * 16 + j * 2 + p]       = f2bf(f.x);
      lB[nt * 512 + (((n + 1) & 15) + h2 * 16) * 16 + j * 2 + p] = f2bf(f.y);
    }
    __syncthreads();

    FragBF af[2], bfr[4];
#pragma unroll
    for (int i = 0; i < 2; ++i) {
      const uint4* p = (const uint4*)&lA[(wm * 2 + i) * 512 + lane * 16];
      af[i].u4[0] = p[0];
      af[i].u4[1] = p[1];
    }
#pragma unroll
    for (int n = 0; n < 4; ++n) {
      const uint4* p = (const uint4*)&lB[(wn * 4 + n) * 512 + lane * 16];
      bfr[n].u4[0] = p[0];
      bfr[n].u4[1] = p[1];
    }
#pragma unroll
    for (int i = 0; i < 2; ++i)
#pragma unroll
      for (int n = 0; n < 4; ++n) acc[i][n] = wmma_bf16(af[i], bfr[n], acc[i][n]);
    __syncthreads();
  }

  // ---- epilogue: C layout lane=(col, hi picks M+8), VGPR g = row ---------
#pragma unroll
  for (int i = 0; i < 2; ++i)
#pragma unroll
    for (int n = 0; n < 4; ++n)
#pragma unroll
      for (int g = 0; g < 8; ++g) {
        float val = acc[i][n][g];
        int m = m0 + wm * 32 + i * 16 + hi * 8 + g;
        int c = n0 + wn * 64 + n * 16 + ln;
        if (MODE == MODE_O) {
          outF[(size_t)m * 1024 + c] = val;
        } else {
          int b = m >> 11, s = m & (Ss - 1);
          int h = c >> 6, d = c & (DHh - 1);
          size_t idx = (((size_t)(b * Hh + h) * Ss) + s) * DHh + d;
          if (MODE == MODE_K || MODE == MODE_V) outF[idx] = val;        // fp32 k/v out
          if (MODE == MODE_Q) outB[idx] = f2bf(val * SCL_LOG2E);        // pre-scaled Q
          if (MODE == MODE_K) outB[idx] = f2bf(val);                    // bf16 K
          if (MODE == MODE_V)                                           // bf16 V^T
            outB[(((size_t)(b * Hh + h) * DHh) + d) * Ss + s] = f2bf(val);
        }
      }
}

// ---------------------------------------------------------------------------
// Pass 1: per-key-column softmax stats m[k], Z[k] over the query axis, in the
// exp2 domain (Q pre-scaled).  128 threads = 4 waves; each wave owns 32 key
// columns, streams q in 16-row WMMA tiles with an online max/sum-exp2.
// ---------------------------------------------------------------------------
__global__ __launch_bounds__(128) void col_softmax_stats(
    const uint16_t* __restrict__ Qbf, const uint16_t* __restrict__ Kbf,
    const uint32_t* __restrict__ maskT, float* __restrict__ mstat,
    float* __restrict__ zstat) {
  const int lane = threadIdx.x & 31;
  const int wv   = threadIdx.x >> 5;
  const int hi   = lane >> 4, ln = lane & 15;
  const int kblk = blockIdx.x & 15;
  const int bh   = blockIdx.x >> 4;
  const int b    = bh >> 4;
  const int k0   = kblk * 128 + wv * 32;

  const uint16_t* Qrow = Qbf + (size_t)bh * Ss * DHh;
  const uint16_t* Krow = Kbf + (size_t)bh * Ss * DHh;
  const uint32_t* mT   = maskT + (size_t)b * 64 * Ss;

  const float NEG = -3.402823466e38f;
  const v8f vzero = {0, 0, 0, 0, 0, 0, 0, 0};

  // B-fragments (K^T): lane = key column, contiguous 32B per dh-step.
  FragBF bk[2][2];
#pragma unroll
  for (int t = 0; t < 2; ++t)
#pragma unroll
    for (int s = 0; s < 2; ++s) {
      const uint16_t* r = Krow + (size_t)(k0 + t * 16 + ln) * DHh + s * 32 + hi * 16;
      bk[t][s].u4[0] = *(const uint4*)r;
      bk[t][s].u4[1] = *(const uint4*)(r + 8);
    }

  float mr[2] = {NEG, NEG};
  float zr[2] = {0.f, 0.f};

  for (int qt = 0; qt < Ss / 16; ++qt) {
    int q0 = qt * 16;
    FragBF aq[2];  // Q A-fragments: lane row q, two 16B chunks per dh-step
#pragma unroll
    for (int s = 0; s < 2; ++s) {
      const uint16_t* r = Qrow + (size_t)(q0 + ln) * DHh + s * 32 + hi * 16;
      aq[s].u4[0] = *(const uint4*)r;
      aq[s].u4[1] = *(const uint4*)(r + 8);
    }
#pragma unroll
    for (int t = 0; t < 2; ++t) {
      v8f sc = vzero;
      sc = wmma_bf16(aq[0], bk[t][0], sc);
      sc = wmma_bf16(aq[1], bk[t][1], sc);
      int k = k0 + t * 16 + ln;
      uint32_t bits = mT[(size_t)(q0 >> 5) * Ss + k] >> ((q0 & 31) + hi * 8);
      float tmax = NEG;
      float vals[8];
#pragma unroll
      for (int g = 0; g < 8; ++g) {
        float v = sc[g];                       // already in exp2 domain
        if ((bits >> g) & 1) v = NEG;
        vals[g] = v;
        tmax = fmaxf(tmax, v);
      }
      tmax = fmaxf(tmax, __shfl_xor(tmax, 16, 32));   // combine M halves
      float tsum = 0.f;
#pragma unroll
      for (int g = 0; g < 8; ++g) tsum += __builtin_exp2f(vals[g] - tmax);
      tsum += __shfl_xor(tsum, 16, 32);
      float mn = fmaxf(mr[t], tmax);                  // online merge
      zr[t] = zr[t] * __builtin_exp2f(mr[t] - mn) + tsum * __builtin_exp2f(tmax - mn);
      mr[t] = mn;
    }
  }
  if (hi == 0) {
#pragma unroll
    for (int t = 0; t < 2; ++t) {
      int k = k0 + t * 16 + ln;
      mstat[(size_t)bh * Ss + k] = mr[t];
      zstat[(size_t)bh * Ss + k] = zr[t];
    }
  }
}

// ---------------------------------------------------------------------------
// Pass 2: hidden[q,d] = sum_k 2^(s[q,k]-m[k])/Z[k] * V[k,d], flash-style.
// 128 threads = 4 waves; wave owns 16 q rows x 64 dh; loops k in 32-tiles.
// K/V tiles are TDM-prefetched into double-buffered LDS (shared by 4 waves);
// score tile -> P (bf16) staged in LDS, re-read as A-fragment for P@V WMMA.
// ---------------------------------------------------------------------------
__global__ __launch_bounds__(128) void attn_pass2(
    const uint16_t* __restrict__ Qbf, const uint16_t* __restrict__ Kbf,
    const uint16_t* __restrict__ Vt, const uint32_t* __restrict__ maskT,
    const float* __restrict__ mstat, const float* __restrict__ zstat,
    float* __restrict__ hidden) {
  __shared__ uint16_t Klds[2][32 * 64];   // double-buffered K tile (4KB each)
  __shared__ uint16_t Vlds[2][64 * 32];   // double-buffered V^T tile (4KB each)
  __shared__ uint16_t Plds[4][16 * 32];   // per-wave 16x32 bf16 P tile

  const int lane = threadIdx.x & 31;
  const int wv   = threadIdx.x >> 5;
  const int hi   = lane >> 4, ln = lane & 15;
  const int qblk = blockIdx.x & 31;
  const int bh   = blockIdx.x >> 5;
  const int b    = bh >> 4, h = bh & 15;
  const int q0   = qblk * 64 + wv * 16;

  const uint16_t* Qrow = Qbf + (size_t)bh * Ss * DHh;
  const uint16_t* Krow = Kbf + (size_t)bh * Ss * DHh;
  const uint16_t* Vrow = Vt + (size_t)bh * DHh * Ss;   // [DH][S] transposed
  const uint32_t* mT   = maskT + ((size_t)b * 64 + (q0 >> 5)) * Ss;
  const float*    mst  = mstat + (size_t)bh * Ss;
  const float*    zst  = zstat + (size_t)bh * Ss;
  const int mshift     = (q0 & 31) + hi * 8;

  const float NEG = -3.402823466e38f;
  const v8f vzero = {0, 0, 0, 0, 0, 0, 0, 0};
  const int NKT = Ss / 32;

  FragBF aq[2];  // Q rows for this wave, loaded once
#pragma unroll
  for (int s = 0; s < 2; ++s) {
    const uint16_t* r = Qrow + (size_t)(q0 + ln) * DHh + s * 32 + hi * 16;
    aq[s].u4[0] = *(const uint4*)r;
    aq[s].u4[1] = *(const uint4*)(r + 8);
  }

  v8f oacc[4] = {vzero, vzero, vzero, vzero};

  // TDM prologue: wave 0 DMAs tile 0 (K: 32 rows x 64, V^T: 64 rows x 32).
  if (wv == 0) {
    tdm_load_2d(lds_off(&Klds[0][0]), Krow, DHh, Ss, DHh, 32, DHh);
    tdm_load_2d(lds_off(&Vlds[0][0]), Vrow, Ss, DHh, 32, DHh, Ss);
  }

  for (int kt = 0; kt < NKT; ++kt) {
    const int k0 = kt * 32;
    const int cb = kt & 1;
    if (wv == 0) {
      if (kt + 1 < NKT) {   // prefetch next tile into the other buffer
        int k1 = k0 + 32;
        tdm_load_2d(lds_off(&Klds[cb ^ 1][0]), Krow + (size_t)k1 * DHh,
                    DHh, Ss, DHh, 32, DHh);
        tdm_load_2d(lds_off(&Vlds[cb ^ 1][0]), Vrow + k1,
                    Ss, DHh, 32, DHh, Ss);
        __builtin_amdgcn_s_wait_tensorcnt(2);   // current pair landed
      } else {
        __builtin_amdgcn_s_wait_tensorcnt(0);
      }
    }
    __syncthreads();   // tile cb visible to all waves

    const uint16_t* kl = &Klds[cb][0];
    const uint16_t* vl = &Vlds[cb][0];

    // ---- scores 16q x 32k from LDS K tile, P = 2^(s-m[k])/Z[k] -> LDS ----
#pragma unroll
    for (int t = 0; t < 2; ++t) {
      FragBF bk0, bk1;
      const uint16_t* r0 = kl + (t * 16 + ln) * DHh + hi * 16;
      bk0.u4[0] = *(const uint4*)r0;
      bk0.u4[1] = *(const uint4*)(r0 + 8);
      const uint16_t* r1 = r0 + 32;
      bk1.u4[0] = *(const uint4*)r1;
      bk1.u4[1] = *(const uint4*)(r1 + 8);
      v8f sc = vzero;
      sc = wmma_bf16(aq[0], bk0, sc);
      sc = wmma_bf16(aq[1], bk1, sc);
      int k = k0 + t * 16 + ln;
      float mcol = mst[k];
      float invz = 1.f / zst[k];
      uint32_t bits = mT[k] >> mshift;
#pragma unroll
      for (int g = 0; g < 8; ++g) {
        float v = sc[g];
        if ((bits >> g) & 1) v = NEG;
        float p = __builtin_exp2f(v - mcol) * invz;  // all-masked col -> 1/S
        Plds[wv][(hi * 8 + g) * 32 + t * 16 + ln] = f2bf(p);
      }
    }
    __syncthreads();
    // ---- P@V: P as A-fragment from LDS, V^T rows from LDS tile -----------
    FragBF pa;
    const uint16_t* pr = &Plds[wv][ln * 32 + hi * 16];
    pa.u4[0] = *(const uint4*)pr;
    pa.u4[1] = *(const uint4*)(pr + 8);
#pragma unroll
    for (int n = 0; n < 4; ++n) {
      FragBF bv;
      const uint16_t* vr = vl + (n * 16 + ln) * 32 + hi * 16;
      bv.u4[0] = *(const uint4*)vr;
      bv.u4[1] = *(const uint4*)(vr + 8);
      oacc[n] = wmma_bf16(pa, bv, oacc[n]);
    }
    __syncthreads();   // done reading buffer cb before it is re-DMA'd
  }

  // ---- write hidden [B,S,H*DH] fp32 (input to the W_o GEMM) --------------
#pragma unroll
  for (int n = 0; n < 4; ++n)
#pragma unroll
    for (int g = 0; g < 8; ++g) {
      int q = q0 + hi * 8 + g;
      int d = h * 64 + n * 16 + ln;
      hidden[((size_t)b * Ss + q) * 1024 + d] = oacc[n][g];
    }
}

// ---------------------------------------------------------------------------
extern "C" void kernel_launch(void* const* d_in, const int* in_sizes, int n_in,
                              void* d_out, int out_size, void* d_ws, size_t ws_size,
                              hipStream_t stream) {
  const float*   query = (const float*)d_in[0];
  const float*   key   = (const float*)d_in[1];
  const float*   value = (const float*)d_in[2];
  const uint8_t* mask  = (const uint8_t*)d_in[3];   // jax bool -> 1 byte
  const float*   W_q   = (const float*)d_in[4];
  const float*   W_k   = (const float*)d_in[5];
  const float*   W_v   = (const float*)d_in[6];
  const float*   W_o   = (const float*)d_in[7];

  float* outF    = (float*)d_out;
  float* hid_out = outF;                                 // [B,S,D]
  float* k_out   = outF + (size_t)Bb * Ss * Dd;          // [B,H,S,DH]
  float* v_out   = k_out + (size_t)Bb * Hh * Ss * DHh;   // [B,H,S,DH]

  uint8_t* ws = (uint8_t*)d_ws;
  const size_t MB = 1024 * 1024;
  uint16_t* Qbf   = (uint16_t*)(ws);                      //  8 MB bf16 [B,H,S,DH] (pre-scaled)
  uint16_t* Kbf   = (uint16_t*)(ws + 8 * MB);             //  8 MB bf16 [B,H,S,DH]
  uint16_t* Vtp   = (uint16_t*)(ws + 16 * MB);            //  8 MB bf16 [B,H,DH,S]
  float*    mstat = (float*)(ws + 24 * MB);               // 256 KB
  float*    zstat = (float*)(ws + 24 * MB + 256 * 1024);  // 256 KB
  uint32_t* maskT = (uint32_t*)(ws + 24 * MB + 512 * 1024); // 1 MB packed mask
  float*    hid   = (float*)(ws + 26 * MB);               // 16 MB fp32 [B,S,H*DH]

  mask_pack<<<(Bb * 64 * Ss) / 256, 256, 0, stream>>>(mask, maskT);

  dim3 gg(4096 / 128, 1024 / 128);
  gemm_proj<MODE_Q><<<gg, 256, 0, stream>>>(query, W_q, nullptr, Qbf);
  gemm_proj<MODE_K><<<gg, 256, 0, stream>>>(key,   W_k, k_out,   Kbf);
  gemm_proj<MODE_V><<<gg, 256, 0, stream>>>(value, W_v, v_out,   Vtp);
  col_softmax_stats<<<Bb * Hh * (Ss / 128), 128, 0, stream>>>(Qbf, Kbf, maskT,
                                                              mstat, zstat);
  attn_pass2<<<Bb * Hh * (Ss / 64), 128, 0, stream>>>(Qbf, Kbf, Vtp, maskT,
                                                      mstat, zstat, hid);
  gemm_proj<MODE_O><<<gg, 256, 0, stream>>>(hid, W_o, hid_out, nullptr);
}
